// CARAFE_59596966199737
// MI455X (gfx1250) — matrix operand order repaired and checked by reading, not compiled, not run on hardware
//
#include <hip/hip_runtime.h>
#include <hip/hip_bf16.h>

typedef __attribute__((ext_vector_type(16))) _Float16 v16h;
typedef __attribute__((ext_vector_type(8)))  float    v8f;

#define C_IN   256
#define C_MID  64
#define ENC    100      // (SCALE*K_UP)^2
#define ENC_P  112      // padded to 7 WMMA M-tiles
#define HW     4096     // 64*64
#define NB     4
#define EPS    1e-5f

// workspace layout (bytes)
#define WS_Y1   0u                       // f16 [NB][C_MID][HW]   = 2,097,152 B
#define WS_Y2   2097152u                 // f32 [NB][ENC][HW]     = 6,553,600 B
#define WS_WC   8650752u                 // f16 [C_MID][C_IN]     =    32,768 B
#define WS_WE   8683520u                 // f16 [ENC_P][576]      =   129,024 B
#define WS_SCB  8812544u                 // f32 [384] folded BN params
// scb: [0..63]=s_comp [64..127]=b_comp [128..227]=s_enc [256..355]=b_enc

// ---------------------------------------------------------------------------
// Prep: pre-cast weights to f16 (zero-pad enc rows >= 100) and fold BN params.
// ---------------------------------------------------------------------------
__global__ void carafe_prep(const float* __restrict__ w_comp,
                            const float* __restrict__ gc, const float* __restrict__ bc,
                            const float* __restrict__ mc, const float* __restrict__ vc,
                            const float* __restrict__ w_enc,
                            const float* __restrict__ ge, const float* __restrict__ be,
                            const float* __restrict__ me, const float* __restrict__ ve,
                            _Float16* __restrict__ wc16, _Float16* __restrict__ we16,
                            float* __restrict__ scb) {
  const int tid = blockIdx.x * blockDim.x + threadIdx.x;
  const int nthr = gridDim.x * blockDim.x;
  for (int t = tid; t < ENC_P * 576; t += nthr) {
    const int m = t / 576, k = t - m * 576;
    we16[t] = (m < ENC) ? (_Float16)w_enc[m * 576 + k] : (_Float16)0.f;
  }
  for (int t = tid; t < C_MID * C_IN; t += nthr)
    wc16[t] = (_Float16)w_comp[t];
  if (tid < C_MID) {
    const float s = gc[tid] * rsqrtf(vc[tid] + EPS);
    scb[tid] = s;
    scb[64 + tid] = bc[tid] - mc[tid] * s;
  }
  if (tid >= 256 && tid < 256 + ENC) {
    const int m = tid - 256;
    const float s = ge[m] * rsqrtf(ve[m] + EPS);
    scb[128 + m] = s;
    scb[256 + m] = be[m] - me[m] * s;
  }
}

// ---------------------------------------------------------------------------
// Kernel 1: 1x1 conv (GEMM 64x256 @ 256x16384) + BN + ReLU.
// One wave32 owns one 16-pixel N-tile and computes the FULL M=64 column:
// one B load feeds 4 WMMAs (x read exactly once). K loop fully unrolled.
// ---------------------------------------------------------------------------
__global__ void carafe_conv1x1_wmma(const float* __restrict__ x,
                                    const _Float16* __restrict__ wc16,
                                    const float* __restrict__ scb,
                                    _Float16* __restrict__ y1) {
  const int lane = threadIdx.x & 31;
  const int ntile = blockIdx.x * (blockDim.x >> 5) + (threadIdx.x >> 5); // 0..1023
  const int n0 = ntile << 4;
  const int b  = n0 >> 12;
  const int p0 = n0 & (HW - 1);
  const int half = lane >> 4;
  const int l16  = lane & 15;

  const float* xb = x + (size_t)b * C_IN * HW + p0 + l16;

  v8f acc[4] = {};
#pragma unroll
  for (int k0 = 0; k0 < C_IN; k0 += 32) {
    v16h B;
#pragma unroll
    for (int e = 0; e < 16; ++e)
      B[e] = (_Float16)xb[(size_t)(k0 + (half << 4) + e) * HW];
#pragma unroll
    for (int mt = 0; mt < 4; ++mt) {
      v16h A;
      const _Float16* ap = wc16 + ((mt << 4) + l16) * C_IN + k0 + (half << 4);
#pragma unroll
      for (int e = 0; e < 8; ++e) { A[e] = ap[e]; A[8 + e] = ap[8 + e]; }
      acc[mt] = __builtin_amdgcn_wmma_f32_16x16x32_f16(false, A, false, B,
                                                       (short)0, acc[mt], false, false);
    }
  }
  // fused BN + ReLU epilogue (folded params), store f16
#pragma unroll
  for (int mt = 0; mt < 4; ++mt)
#pragma unroll
    for (int v = 0; v < 8; ++v) {
      const int m = (mt << 4) + v + (half << 3);
      float r = acc[mt][v] * scb[m] + scb[64 + m];
      r = r > 0.f ? r : 0.f;
      y1[((size_t)b * C_MID + m) * HW + p0 + l16] = (_Float16)r;
    }
}

// ---------------------------------------------------------------------------
// Kernel 2: 3x3 conv (im2col GEMM 112x576 @ 576x16384) + BN.
// One wave32 owns one N-tile, full M column (7 accumulators): one im2col
// B load feeds 7 WMMAs. K loop fully unrolled -> im2col decode is constant.
// ---------------------------------------------------------------------------
__global__ void carafe_conv3x3_wmma(const _Float16* __restrict__ y1,
                                    const _Float16* __restrict__ we16,
                                    const float* __restrict__ scb,
                                    float* __restrict__ y2) {
  const int lane = threadIdx.x & 31;
  const int ntile = blockIdx.x * (blockDim.x >> 5) + (threadIdx.x >> 5); // 0..1023
  const int n0 = ntile << 4;
  const int b  = n0 >> 12;
  const int p0 = n0 & (HW - 1);
  const int half = lane >> 4;
  const int l16  = lane & 15;

  const int p  = p0 + l16;         // 16 pixels share row h (64 % 16 == 0)
  const int h  = p >> 6;
  const int wc = p & 63;
  const _Float16* yb = y1 + (size_t)b * C_MID * HW;

  v8f acc[7] = {};
#pragma unroll
  for (int k0 = 0; k0 < 576; k0 += 32) {
    v16h B;
#pragma unroll
    for (int e = 0; e < 16; ++e) {
      const int kB = k0 + (half << 4) + e;   // k = c*9 + ki*3 + kj  (const after unroll)
      const int c  = kB / 9;
      const int r9 = kB - c * 9;
      const int ki = r9 / 3;
      const int kj = r9 - ki * 3;
      const int hh = h + ki - 1;
      const int ww = wc + kj - 1;
      _Float16 v = (_Float16)0.f;
      if ((unsigned)hh < 64u && (unsigned)ww < 64u)
        v = yb[((size_t)c << 12) + (hh << 6) + ww];
      B[e] = v;
    }
#pragma unroll
    for (int mt = 0; mt < 7; ++mt) {
      v16h A;
      const _Float16* ap = we16 + ((mt << 4) + l16) * 576 + k0 + (half << 4);
#pragma unroll
      for (int e = 0; e < 8; ++e) { A[e] = ap[e]; A[8 + e] = ap[8 + e]; }
      acc[mt] = __builtin_amdgcn_wmma_f32_16x16x32_f16(false, A, false, B,
                                                       (short)0, acc[mt], false, false);
    }
  }
#pragma unroll
  for (int mt = 0; mt < 7; ++mt)
#pragma unroll
    for (int v = 0; v < 8; ++v) {
      const int m = (mt << 4) + v + (half << 3);
      if (m < ENC)
        y2[((size_t)b * ENC + m) * HW + p0 + l16] =
            acc[mt][v] * scb[128 + m] + scb[256 + m];
    }
}

// ---------------------------------------------------------------------------
// Kernel 3: pixel-shuffle + softmax(25) + 5x5 reassembly on 2x grid.
// Block = 256 threads = one 16x16 output tile. Channels staged 8-at-a-time
// through LDS (12x12 x-patch per channel, ~25x reuse). Bandwidth-bound.
// ---------------------------------------------------------------------------
#define CH_BLK 8
__global__ void carafe_reassemble(const float* __restrict__ x,
                                  const float* __restrict__ y2,
                                  float* __restrict__ out) {
  __shared__ float patch[CH_BLK][12][12];

  const int tid = threadIdx.x;
  const int ty = tid >> 4, tx = tid & 15;
  const int b  = blockIdx.z;
  const int i0 = blockIdx.y << 4;
  const int j0 = blockIdx.x << 4;
  const int i = i0 + ty, j = j0 + tx;
  const int h = i >> 1, w = j >> 1;
  const int off = ((i & 1) << 1) + (j & 1);     // pixel-shuffle sub-position
  const int xh0 = (i0 >> 1) - 2;
  const int xw0 = (j0 >> 1) - 2;

  // softmax over the 25 kernel logits for this output pixel
  float wk[25];
  {
    const float* lp = y2 + ((size_t)b * ENC + off) * HW + (h << 6) + w;
    float mx = -3.4e38f;
#pragma unroll
    for (int k = 0; k < 25; ++k) {
      wk[k] = lp[(size_t)(k << 2) * HW];        // channel k*4 + off
      mx = wk[k] > mx ? wk[k] : mx;
    }
    float sum = 0.f;
#pragma unroll
    for (int k = 0; k < 25; ++k) { wk[k] = __expf(wk[k] - mx); sum += wk[k]; }
    const float inv = 1.f / sum;
#pragma unroll
    for (int k = 0; k < 25; ++k) wk[k] *= inv;
  }

  const int lr = ty >> 1;
  const int lc = tx >> 1;

  for (int cb = 0; cb < C_IN; cb += CH_BLK) {
    __syncthreads();
    for (int t = tid; t < CH_BLK * 144; t += 256) {
      const int ch = t / 144;
      const int rem = t - ch * 144;
      const int rr = rem / 12;
      const int cc = rem - rr * 12;
      const int gh = xh0 + rr, gw = xw0 + cc;
      float v = 0.f;
      if ((unsigned)gh < 64u && (unsigned)gw < 64u)
        v = x[(((size_t)b * C_IN + cb + ch) << 12) + (gh << 6) + gw];
      patch[ch][rr][cc] = v;
    }
    // prefetch next channel block into caches while we compute
    if (cb + CH_BLK < C_IN && tid < CH_BLK)
      __builtin_prefetch(&x[(((size_t)b * C_IN + cb + CH_BLK + tid) << 12)], 0, 0);
    __syncthreads();

#pragma unroll
    for (int ch = 0; ch < CH_BLK; ++ch) {
      float s = 0.f;
#pragma unroll
      for (int ki = 0; ki < 5; ++ki)
#pragma unroll
        for (int kj = 0; kj < 5; ++kj)
          s = fmaf(wk[ki * 5 + kj], patch[ch][lr + ki][lc + kj], s);
      out[(((size_t)b * C_IN + cb + ch) << 14) + (i << 7) + j] = s;
    }
  }
}

// ---------------------------------------------------------------------------
extern "C" void kernel_launch(void* const* d_in, const int* in_sizes, int n_in,
                              void* d_out, int out_size, void* d_ws, size_t ws_size,
                              hipStream_t stream) {
  const float* x          = (const float*)d_in[0];
  const float* w_comp     = (const float*)d_in[1];
  const float* gamma_comp = (const float*)d_in[2];
  const float* beta_comp  = (const float*)d_in[3];
  const float* mean_comp  = (const float*)d_in[4];
  const float* var_comp   = (const float*)d_in[5];
  const float* w_enc      = (const float*)d_in[6];
  const float* gamma_enc  = (const float*)d_in[7];
  const float* beta_enc   = (const float*)d_in[8];
  const float* mean_enc   = (const float*)d_in[9];
  const float* var_enc    = (const float*)d_in[10];
  float* out = (float*)d_out;

  char* ws = (char*)d_ws;
  _Float16* y1   = (_Float16*)(ws + WS_Y1);
  float*    y2   = (float*)   (ws + WS_Y2);
  _Float16* wc16 = (_Float16*)(ws + WS_WC);
  _Float16* we16 = (_Float16*)(ws + WS_WE);
  float*    scb  = (float*)   (ws + WS_SCB);

  carafe_prep<<<dim3(64), dim3(256), 0, stream>>>(
      w_comp, gamma_comp, beta_comp, mean_comp, var_comp,
      w_enc, gamma_enc, beta_enc, mean_enc, var_enc, wc16, we16, scb);

  // conv1x1: 1024 N-tile waves, 8 waves/block
  carafe_conv1x1_wmma<<<dim3(128), dim3(256), 0, stream>>>(x, wc16, scb, y1);

  // conv3x3: 1024 N-tile waves, 8 waves/block
  carafe_conv3x3_wmma<<<dim3(128), dim3(256), 0, stream>>>(y1, we16, scb, y2);

  // reassembly: one 16x16 tile per block over 128x128 grid, 4 batches
  carafe_reassemble<<<dim3(8, 8, NB), dim3(256), 0, stream>>>(x, y2, out);
}
